// Network_72404558676634
// MI455X (gfx1250) — matrix operand order repaired and checked by reading, not compiled, not run on hardware
//
#include <hip/hip_runtime.h>
#include <hip/hip_bf16.h>
#include <stdint.h>

typedef __attribute__((ext_vector_type(8)))  _Float16 v8h;
typedef __attribute__((ext_vector_type(16))) _Float16 v16h;
typedef __attribute__((ext_vector_type(8)))  float    v8f;

#define LAYERS      16
#define SUBLAYERS   12
#define NPLANES     192
#define RAYS        1024
#define NPTS        (NPLANES * RAYS)      // 196608 rows through the MLPs
#define HIDDEN      384
#define BH          64
#define IN_H        512
#define IN_W        768
#define POS_LEVEL   10
#define DEPTH_LEVEL 8
#define ENC_RAW     56
#define ENC_K       64                    // 56 padded to 64 (K multiple of 32)
#define VENC_K      32                    // 12 padded to 32
#define OUT_N       32                    // 25 padded to 32
#define MTILE       64                    // rows per mlp_main block (4 subtiles)
#define PI_HALF     1.57079632679489661923f

__device__ __forceinline__ v16h cat8(v8h lo, v8h hi) {
  return __builtin_shufflevector(lo, hi, 0,1,2,3,4,5,6,7,8,9,10,11,12,13,14,15);
}
__device__ __forceinline__ float fsigmoid(float x) { return 1.0f / (1.0f + __expf(-x)); }
__device__ __forceinline__ float fleaky(float x)   { return x >= 0.0f ? x : 0.01f * x; }

__device__ __forceinline__ void mat3mul(const float* A, const float* B, float* C) {
  #pragma unroll
  for (int i = 0; i < 3; ++i)
    #pragma unroll
    for (int j = 0; j < 3; ++j)
      C[i*3+j] = A[i*3+0]*B[0+j] + A[i*3+1]*B[3+j] + A[i*3+2]*B[6+j];
}

// ---------------------------------------------------------------------------
// Kernel 1: fp32 weights -> f16, transposed to N-major WT[n][k] (WMMA B layout
// friendly: each lane reads 16 contiguous K values of its column). Also packs
// biases (padded) into fp32 arrays.
// ---------------------------------------------------------------------------
__global__ void prep_weights(
    const float* __restrict__ w0,  const float* __restrict__ b0,
    const float* __restrict__ w1,  const float* __restrict__ b1,
    const float* __restrict__ w2,  const float* __restrict__ b2,
    const float* __restrict__ w3,  const float* __restrict__ b3,
    const float* __restrict__ w4,  const float* __restrict__ b4,
    const float* __restrict__ wo,  const float* __restrict__ bo,
    const float* __restrict__ bw0, const float* __restrict__ bb0,
    const float* __restrict__ bw1, const float* __restrict__ bb1,
    const float* __restrict__ bwo, const float* __restrict__ bbo,
    _Float16* __restrict__ wt0,   // [384][64]
    _Float16* __restrict__ wth,   // 4 x [384][384]
    _Float16* __restrict__ wto,   // [32][384]
    _Float16* __restrict__ bwt0,  // [64][32]
    _Float16* __restrict__ bwt1,  // [64][64]
    _Float16* __restrict__ bwto,  // [16][64]
    float* __restrict__ bias,     // [5*384 + 32]
    float* __restrict__ bbias)    // [64 + 64 + 16]
{
  int i = blockIdx.x * blockDim.x + threadIdx.x;
  if (i < HIDDEN * ENC_K) {                         // WT0
    int n = i / ENC_K, k = i % ENC_K;
    wt0[i] = (_Float16)((k < ENC_RAW) ? w0[k * HIDDEN + n] : 0.0f);
    return;
  }
  i -= HIDDEN * ENC_K;
  if (i < 4 * HIDDEN * HIDDEN) {                    // WT1..WT4
    int l = i / (HIDDEN * HIDDEN);
    int j = i % (HIDDEN * HIDDEN);
    int n = j / HIDDEN, k = j % HIDDEN;
    const float* w = (l == 0) ? w1 : (l == 1) ? w2 : (l == 2) ? w3 : w4;
    wth[(size_t)l * HIDDEN * HIDDEN + (size_t)n * HIDDEN + k] = (_Float16)w[k * HIDDEN + n];
    return;
  }
  i -= 4 * HIDDEN * HIDDEN;
  if (i < OUT_N * HIDDEN) {                         // WTo (pad N 25->32)
    int n = i / HIDDEN, k = i % HIDDEN;
    wto[i] = (_Float16)((n < 25) ? wo[k * 25 + n] : 0.0f);
    return;
  }
  i -= OUT_N * HIDDEN;
  if (i < BH * VENC_K) {                            // BWT0 (pad K 12->32)
    int n = i / VENC_K, k = i % VENC_K;
    bwt0[i] = (_Float16)((k < 12) ? bw0[k * BH + n] : 0.0f);
    return;
  }
  i -= BH * VENC_K;
  if (i < BH * BH) {                                // BWT1
    int n = i / BH, k = i % BH;
    bwt1[i] = (_Float16)bw1[k * BH + n];
    return;
  }
  i -= BH * BH;
  if (i < 16 * BH) {                                // BWTo (pad N 8->16)
    int n = i / BH, k = i % BH;
    bwto[i] = (_Float16)((n < 8) ? bwo[k * 8 + n] : 0.0f);
    return;
  }
  i -= 16 * BH;
  if (i < 5 * HIDDEN + OUT_N) {                     // packed main biases
    float v;
    if      (i < 384)  v = b0[i];
    else if (i < 768)  v = b1[i - 384];
    else if (i < 1152) v = b2[i - 768];
    else if (i < 1536) v = b3[i - 1152];
    else if (i < 1920) v = b4[i - 1536];
    else { int j = i - 1920; v = (j < 25) ? bo[j] : 0.0f; }
    bias[i] = v;
    return;
  }
  i -= 5 * HIDDEN + OUT_N;
  if (i < 144) {                                    // packed basis biases
    float v;
    if      (i < 64)  v = bb0[i];
    else if (i < 128) v = bb1[i - 64];
    else { int j = i - 128; v = (j < 8) ? bbo[j] : 0.0f; }
    bbias[i] = v;
  }
}

// ---------------------------------------------------------------------------
// Kernel 2: per-plane homographies Hs[p] (3x3), plus Ki = inv(feat_k) and
// camera center into misc[0..8]=Ki, misc[9..11]=camera.
// ---------------------------------------------------------------------------
__global__ void prep_planes(
    const float* __restrict__ ref_rT, const float* __restrict__ ref_t,
    const float* __restrict__ ref_k,  const float* __restrict__ feat_r,
    const float* __restrict__ feat_t, const float* __restrict__ feat_center,
    const float* __restrict__ feat_k, const float* __restrict__ planes,
    float* __restrict__ Hs, float* __restrict__ misc)
{
  float Ki[9];
  {
    float a = feat_k[0], b = feat_k[1], c = feat_k[2];
    float d = feat_k[3], e = feat_k[4], f = feat_k[5];
    float g = feat_k[6], h = feat_k[7], ii = feat_k[8];
    float A =  (e*ii - f*h), B = -(d*ii - f*g), C =  (d*h - e*g);
    float inv = 1.0f / (a*A + b*B + c*C);
    Ki[0] = A*inv;             Ki[1] = -(b*ii - c*h)*inv;  Ki[2] =  (b*f - c*e)*inv;
    Ki[3] = B*inv;             Ki[4] =  (a*ii - c*g)*inv;  Ki[5] = -(a*f - c*d)*inv;
    Ki[6] = C*inv;             Ki[7] = -(a*h - b*g)*inv;   Ki[8] =  (a*e - b*d)*inv;
  }
  float newR[9]; mat3mul(feat_r, ref_rT, newR);
  float newT[3];
  #pragma unroll
  for (int i = 0; i < 3; ++i)
    newT[i] = -(newR[i*3]*ref_t[0] + newR[i*3+1]*ref_t[1] + newR[i*3+2]*ref_t[2]) + feat_t[i];
  float Ha[9];
  #pragma unroll
  for (int i = 0; i < 3; ++i)
    #pragma unroll
    for (int j = 0; j < 3; ++j)
      Ha[i*3+j] = newR[j*3+i];
  float hv[3];
  #pragma unroll
  for (int i = 0; i < 3; ++i)
    hv[i] = Ha[i*3]*newT[0] + Ha[i*3+1]*newT[1] + Ha[i*3+2]*newT[2];
  const float Hc = hv[2];

  int p = threadIdx.x;
  if (p < NPLANES) {
    float s = 1.0f / (-planes[p] - Hc);
    float M[9];
    #pragma unroll
    for (int i = 0; i < 3; ++i)
      #pragma unroll
      for (int j = 0; j < 3; ++j)
        M[i*3+j] = Ha[i*3+j] + hv[i] * Ha[6+j] * s;
    float T1[9]; mat3mul(ref_k, M, T1);
    float Hp[9]; mat3mul(T1, Ki, Hp);
    #pragma unroll
    for (int j = 0; j < 9; ++j) Hs[p*9 + j] = Hp[j];
  }
  if (threadIdx.x == 0) {
    #pragma unroll
    for (int j = 0; j < 9; ++j) misc[j] = Ki[j];
    #pragma unroll
    for (int i = 0; i < 3; ++i)
      misc[9 + i] = ref_rT[0*3+i]*feat_center[0] + ref_rT[1*3+i]*feat_center[1]
                  + ref_rT[2*3+i]*feat_center[2] + ref_t[i];
  }
}

// ---------------------------------------------------------------------------
// Kernel 3: per-point warp, positional/view encodings (f16, padded), and the
// trilinear grid-sample of sigmoid(mpi_c).
// ---------------------------------------------------------------------------
__global__ __launch_bounds__(256) void prep_points(
    const float* __restrict__ Hs, const float* __restrict__ misc,
    const float* __restrict__ planes, const int* __restrict__ selection,
    const float* __restrict__ mpi_c,
    _Float16* __restrict__ enc,   // [NPTS][64]
    _Float16* __restrict__ venc,  // [NPTS][32]
    float* __restrict__ rgbbuf)   // [NPTS][3]
{
  int idx = blockIdx.x * blockDim.x + threadIdx.x;
  if (idx >= NPTS) return;
  const int p = idx / RAYS;
  const int r = idx - p * RAYS;
  const int sel = selection[r];
  const float cx = (float)(sel % IN_W);
  const float cy = (float)(sel / IN_W);

  const float* H = Hs + p * 9;
  float p0 = H[0]*cx + H[1]*cy + H[2];
  float p1 = H[3]*cx + H[4]*cy + H[5];
  float p2 = H[6]*cx + H[7]*cy + H[8];
  float inv2 = 1.0f / p2;
  float rx = p0 * inv2, ry = p1 * inv2;              // ref_coords
  float wx = rx / (IN_W - 1.0f) * 2.0f - 1.0f;       // warp
  float wy = ry / (IN_H - 1.0f) * 2.0f - 1.0f;

  // ---- positional encoding: [sin(vxyz), cos(vxyz)], vxyz len 28 ----
  float vv[2*POS_LEVEL + DEPTH_LEVEL];
  float f = PI_HALF;
  #pragma unroll
  for (int l = 0; l < POS_LEVEL; ++l) { vv[l] = wx*f; vv[POS_LEVEL + l] = wy*f; f *= 2.0f; }
  float z = 2.0f * p / (NPLANES - 1.0f) - 1.0f;
  f = PI_HALF;
  #pragma unroll
  for (int l = 0; l < DEPTH_LEVEL; ++l) { vv[2*POS_LEVEL + l] = z*f; f *= 2.0f; }
  _Float16* e = enc + (size_t)idx * ENC_K;
  #pragma unroll
  for (int i = 0; i < 28; ++i) {
    e[i]      = (_Float16)__sinf(vv[i]);
    e[28 + i] = (_Float16)__cosf(vv[i]);
  }
  #pragma unroll
  for (int i = ENC_RAW; i < ENC_K; ++i) e[i] = (_Float16)0.0f;

  // ---- view-direction encoding ----
  const float* Ki  = misc;
  const float* cam = misc + 9;
  float pl = planes[p];
  float c3x = rx * pl, c3y = ry * pl, c3z = pl;
  float X = Ki[0]*c3x + Ki[1]*c3y + Ki[2]*c3z - cam[0];
  float Y = Ki[3]*c3x + Ki[4]*c3y + Ki[5]*c3z - cam[1];
  float Z = Ki[6]*c3x + Ki[7]*c3y + Ki[8]*c3z - cam[2];
  float nrm = sqrtf(X*X + Y*Y + Z*Z) + 1e-7f;
  float vx = X / nrm, vy = Y / nrm;
  float hvv[6] = { vx, 2.0f*vx, 4.0f*vx, vy, 2.0f*vy, 4.0f*vy };
  _Float16* ve = venc + (size_t)idx * VENC_K;
  #pragma unroll
  for (int i = 0; i < 6; ++i) {
    ve[i]     = (_Float16)__sinf(PI_HALF * hvv[i]);
    ve[6 + i] = (_Float16)__cosf(PI_HALF * hvv[i]);
  }
  #pragma unroll
  for (int i = 12; i < VENC_K; ++i) ve[i] = (_Float16)0.0f;

  // ---- trilinear grid sample of sigmoid(mpi_c): vol[l][c][y][x] ----
  float gx = rx, gy = ry;
  float gz = (float)(p / SUBLAYERS);                 // exact: (depth+1)/2*(L-1)
  float x0 = floorf(gx), y0 = floorf(gy), z0 = floorf(gz);
  float out3[3] = {0.0f, 0.0f, 0.0f};
  #pragma unroll
  for (int dz = 0; dz < 2; ++dz)
    #pragma unroll
    for (int dy = 0; dy < 2; ++dy)
      #pragma unroll
      for (int dx = 0; dx < 2; ++dx) {
        float xi = x0 + dx, yi = y0 + dy, zi = z0 + dz;
        bool valid = (xi >= 0.0f) & (xi < (float)IN_W) & (yi >= 0.0f) &
                     (yi < (float)IN_H) & (zi >= 0.0f) & (zi < (float)LAYERS);
        if (valid) {
          float w = (1.0f - fabsf(gx - xi)) * (1.0f - fabsf(gy - yi)) * (1.0f - fabsf(gz - zi));
          int xc = (int)xi, yc = (int)yi, zc = (int)zi;
          size_t base = (((size_t)zc * 3) * IN_H + yc) * IN_W + xc;
          #pragma unroll
          for (int c = 0; c < 3; ++c)
            out3[c] += w * fsigmoid(mpi_c[base + (size_t)c * IN_H * IN_W]);
        }
      }
  rgbbuf[(size_t)idx * 3 + 0] = out3[0];
  rgbbuf[(size_t)idx * 3 + 1] = out3[1];
  rgbbuf[(size_t)idx * 3 + 2] = out3[2];
}

// ---------------------------------------------------------------------------
// Kernel 4: fused main MLP (56->384 ->384 x4 ->25) in f16 WMMA, f32 accum.
// 256 threads = 8 waves; 64-row tile per block (4 M-subtiles per wave, so each
// B fragment load feeds 4 WMMAs -> 4x less L2 weight traffic than a 16-row
// tile). Single in-place LDS activation buffer (50 KB); two barriers/layer:
// all K-loop reads finish before the in-place overwrite.
// ---------------------------------------------------------------------------
__global__ __launch_bounds__(256) void mlp_main(
    const _Float16* __restrict__ enc,   // [NPTS][64]
    const _Float16* __restrict__ wt0,   // [384][64]
    const _Float16* __restrict__ wth,   // 4 x [384][384]
    const _Float16* __restrict__ wto,   // [32][384]
    const float* __restrict__ bias,     // [5*384 + 32]
    float* __restrict__ outbuf)         // [NPTS][32]
{
  constexpr int ASTR = HIDDEN + 8;      // f16 stride, keeps 16B row alignment
  __shared__ _Float16 act[MTILE * ASTR];
  const int tid  = threadIdx.x;
  const int lane = tid & 31;
  const int wave = tid >> 5;            // 0..7
  const int half = lane >> 4;           // hi/lo lane group
  const int l16  = lane & 15;
  const size_t row0 = (size_t)blockIdx.x * MTILE;

  // stage MTILE x 64 f16 encoding tile
  for (int i = tid; i < MTILE * (ENC_K / 2); i += 256) {
    int rr = i >> 5, cc = i & 31;
    ((uint32_t*)&act[rr * ASTR])[cc] =
        ((const uint32_t*)(enc + (row0 + rr) * ENC_K))[cc];
  }
  __syncthreads();

  for (int layer = 0; layer < 5; ++layer) {
    const int K = (layer == 0) ? ENC_K : HIDDEN;
    const _Float16* WT = (layer == 0) ? wt0 : (wth + (size_t)(layer - 1) * HIDDEN * HIDDEN);
    const float* bl = bias + layer * HIDDEN;
    const int nb0 = wave * 48;

    float bv0 = bl[nb0 + l16], bv1 = bl[nb0 + 16 + l16], bv2 = bl[nb0 + 32 + l16];
    v8f s0 = {bv0, bv0, bv0, bv0, bv0, bv0, bv0, bv0};
    v8f s1 = {bv1, bv1, bv1, bv1, bv1, bv1, bv1, bv1};
    v8f s2 = {bv2, bv2, bv2, bv2, bv2, bv2, bv2, bv2};
    v8f acc[4][3];
    #pragma unroll
    for (int ms = 0; ms < 4; ++ms) { acc[ms][0] = s0; acc[ms][1] = s1; acc[ms][2] = s2; }

    for (int k0 = 0; k0 < K; k0 += 32) {
      const int kb = k0 + (half ? 16 : 0);
      v16h A[4];
      #pragma unroll
      for (int ms = 0; ms < 4; ++ms) {
        const _Float16* arow = &act[(ms * 16 + l16) * ASTR + kb];
        A[ms] = cat8(*(const v8h*)arow, *(const v8h*)(arow + 8));
      }
      const _Float16* wrow = WT + (size_t)(nb0 + l16) * K + k0 + (half ? 16 : 0);
      v16h B0 = *(const v16h*)wrow;
      v16h B1 = *(const v16h*)(wrow + (size_t)16 * K);
      v16h B2 = *(const v16h*)(wrow + (size_t)32 * K);
      if (k0 + 32 < K) __builtin_prefetch((const void*)(wrow + 32), 0, 1);
      #pragma unroll
      for (int ms = 0; ms < 4; ++ms) {
        acc[ms][0] = __builtin_amdgcn_wmma_f32_16x16x32_f16(false, A[ms], false, B0, (short)0, acc[ms][0], false, false);
        acc[ms][1] = __builtin_amdgcn_wmma_f32_16x16x32_f16(false, A[ms], false, B1, (short)0, acc[ms][1], false, false);
        acc[ms][2] = __builtin_amdgcn_wmma_f32_16x16x32_f16(false, A[ms], false, B2, (short)0, acc[ms][2], false, false);
      }
    }

    __syncthreads();                    // every wave done reading act
    #pragma unroll
    for (int ms = 0; ms < 4; ++ms)
      #pragma unroll
      for (int j = 0; j < 8; ++j) {
        int m = ms * 16 + j + (half ? 8 : 0);
        act[m * ASTR + nb0 +      l16] = (_Float16)fleaky(acc[ms][0][j]);
        act[m * ASTR + nb0 + 16 + l16] = (_Float16)fleaky(acc[ms][1][j]);
        act[m * ASTR + nb0 + 32 + l16] = (_Float16)fleaky(acc[ms][2][j]);
      }
    __syncthreads();                    // writes visible before next layer
  }

  // output layer: 32 cols (25 real); wave -> (M-subtile, 16-col tile)
  {
    const int ms = wave >> 1;
    const int nb = (wave & 1) * 16;
    float bv = bias[5 * HIDDEN + nb + l16];
    v8f acc = {bv, bv, bv, bv, bv, bv, bv, bv};
    for (int k0 = 0; k0 < HIDDEN; k0 += 32) {
      const int kb = k0 + (half ? 16 : 0);
      const _Float16* arow = &act[(ms * 16 + l16) * ASTR + kb];
      v16h A = cat8(*(const v8h*)arow, *(const v8h*)(arow + 8));
      const _Float16* wrow = wto + (size_t)(nb + l16) * HIDDEN + k0 + (half ? 16 : 0);
      v16h B = *(const v16h*)wrow;
      acc = __builtin_amdgcn_wmma_f32_16x16x32_f16(false, A, false, B, (short)0, acc, false, false);
    }
    #pragma unroll
    for (int j = 0; j < 8; ++j) {
      int m = ms * 16 + j + (half ? 8 : 0);
      outbuf[(row0 + m) * OUT_N + nb + l16] = acc[j];
    }
  }
}

// ---------------------------------------------------------------------------
// Kernel 5: fused basis MLP (12->64->64->8, tanh out) in f16 WMMA.
// 128 threads = 4 waves; 16-row tile per block; 16 N columns per wave.
// ---------------------------------------------------------------------------
__global__ __launch_bounds__(128) void mlp_basis(
    const _Float16* __restrict__ venc,  // [NPTS][32]
    const _Float16* __restrict__ bwt0,  // [64][32]
    const _Float16* __restrict__ bwt1,  // [64][64]
    const _Float16* __restrict__ bwto,  // [16][64]
    const float* __restrict__ bbias,    // [64 + 64 + 16]
    float* __restrict__ basisbuf)       // [NPTS][8]
{
  constexpr int BSTR = BH + 8;
  __shared__ _Float16 act[2][16 * BSTR];
  const int tid  = threadIdx.x;
  const int lane = tid & 31;
  const int wave = tid >> 5;            // 0..3
  const int half = lane >> 4;
  const int l16  = lane & 15;
  const size_t row0 = (size_t)blockIdx.x * 16;

  for (int i = tid; i < 16 * (VENC_K / 2); i += 128) {
    int rr = i / (VENC_K / 2), cc = i % (VENC_K / 2);
    ((uint32_t*)&act[0][rr * BSTR])[cc] =
        ((const uint32_t*)(venc + (row0 + rr) * VENC_K))[cc];
  }
  __syncthreads();

  int cur = 0;
  #pragma unroll
  for (int layer = 0; layer < 2; ++layer) {
    const int K = (layer == 0) ? VENC_K : BH;
    const _Float16* WT = (layer == 0) ? bwt0 : bwt1;
    const float* bl = bbias + layer * BH;
    const int nb = wave * 16;
    float bv = bl[nb + l16];
    v8f acc = {bv, bv, bv, bv, bv, bv, bv, bv};
    for (int k0 = 0; k0 < K; k0 += 32) {
      const int kb = k0 + (half ? 16 : 0);
      const _Float16* arow = &act[cur][l16 * BSTR + kb];
      v16h A = cat8(*(const v8h*)arow, *(const v8h*)(arow + 8));
      const _Float16* wrow = WT + (size_t)(nb + l16) * K + k0 + (half ? 16 : 0);
      v16h B = *(const v16h*)wrow;
      acc = __builtin_amdgcn_wmma_f32_16x16x32_f16(false, A, false, B, (short)0, acc, false, false);
    }
    _Float16* nxt = act[cur ^ 1];
    #pragma unroll
    for (int j = 0; j < 8; ++j) {
      int m = j + (half ? 8 : 0);
      nxt[m * BSTR + nb + l16] = (_Float16)fleaky(acc[j]);
    }
    __syncthreads();
    cur ^= 1;
  }

  if (wave == 0) {                      // 16 output cols (8 real), tanh
    float bv = bbias[128 + l16];
    v8f acc = {bv, bv, bv, bv, bv, bv, bv, bv};
    for (int k0 = 0; k0 < BH; k0 += 32) {
      const int kb = k0 + (half ? 16 : 0);
      const _Float16* arow = &act[cur][l16 * BSTR + kb];
      v16h A = cat8(*(const v8h*)arow, *(const v8h*)(arow + 8));
      const _Float16* wrow = bwto + (size_t)l16 * BH + k0 + (half ? 16 : 0);
      v16h B = *(const v16h*)wrow;
      acc = __builtin_amdgcn_wmma_f32_16x16x32_f16(false, A, false, B, (short)0, acc, false, false);
    }
    if (l16 < 8) {
      #pragma unroll
      for (int j = 0; j < 8; ++j) {
        int m = j + (half ? 8 : 0);
        basisbuf[(row0 + m) * 8 + l16] = tanhf(acc[j]);
      }
    }
  }
}

// ---------------------------------------------------------------------------
// Kernel 6: alpha compositing over planes (exclusive cumprod), per ray.
// ---------------------------------------------------------------------------
__global__ __launch_bounds__(256) void composite(
    const float* __restrict__ outbuf,   // [NPTS][32]
    const float* __restrict__ rgbbuf,   // [NPTS][3]
    const float* __restrict__ basisbuf, // [NPTS][8]
    float* __restrict__ d_out)          // [3][RAYS]
{
  int r = blockIdx.x * blockDim.x + threadIdx.x;
  if (r >= RAYS) return;
  float T = 1.0f;
  float comp[3] = {0.0f, 0.0f, 0.0f};
  float coeff[3][8];
  for (int p = 0; p < NPLANES; ++p) {
    size_t row = (size_t)p * RAYS + r;
    if ((p % SUBLAYERS) == 0) {
      const float* o = outbuf + ((size_t)(p / SUBLAYERS) * SUBLAYERS * RAYS + r) * OUT_N;
      #pragma unroll
      for (int c = 0; c < 3; ++c)
        #pragma unroll
        for (int k = 0; k < 8; ++k)
          coeff[c][k] = tanhf(o[1 + c * 8 + k]);
    }
    float alpha = fsigmoid(outbuf[row * OUT_N] - 5.0f);
    const float* bs = basisbuf + row * 8;
    const float* rg = rgbbuf + row * 3;
    float w = T * alpha;
    #pragma unroll
    for (int c = 0; c < 3; ++c) {
      float il = 0.0f;
      #pragma unroll
      for (int k = 0; k < 8; ++k) il += coeff[c][k] * bs[k];
      float v = rg[c] + il;
      v = fminf(fmaxf(v, 0.0f), 1.0f);
      comp[c] += w * v;
    }
    T *= (1.0f - alpha);
  }
  d_out[0 * RAYS + r] = comp[0];
  d_out[1 * RAYS + r] = comp[1];
  d_out[2 * RAYS + r] = comp[2];
}

// ---------------------------------------------------------------------------
extern "C" void kernel_launch(void* const* d_in, const int* in_sizes, int n_in,
                              void* d_out, int out_size, void* d_ws, size_t ws_size,
                              hipStream_t stream) {
  const float* mpi_c = (const float*)d_in[0];
  const float* w0 = (const float*)d_in[1];   const float* b0 = (const float*)d_in[2];
  const float* w1 = (const float*)d_in[3];   const float* b1 = (const float*)d_in[4];
  const float* w2 = (const float*)d_in[5];   const float* b2 = (const float*)d_in[6];
  const float* w3 = (const float*)d_in[7];   const float* b3 = (const float*)d_in[8];
  const float* w4 = (const float*)d_in[9];   const float* b4 = (const float*)d_in[10];
  const float* wo = (const float*)d_in[11];  const float* bo = (const float*)d_in[12];
  const float* bw0 = (const float*)d_in[13]; const float* bb0 = (const float*)d_in[14];
  const float* bw1 = (const float*)d_in[15]; const float* bb1 = (const float*)d_in[16];
  const float* bwo = (const float*)d_in[17]; const float* bbo = (const float*)d_in[18];
  const float* ref_rT = (const float*)d_in[19];
  const float* ref_t  = (const float*)d_in[20];
  const float* ref_k  = (const float*)d_in[21];
  const float* feat_r = (const float*)d_in[22];
  const float* feat_t = (const float*)d_in[23];
  const float* feat_center = (const float*)d_in[24];
  const float* feat_k = (const float*)d_in[25];
  const float* planes = (const float*)d_in[26];
  const int*   selection = (const int*)d_in[27];
  (void)in_sizes; (void)n_in; (void)out_size; (void)ws_size;

  char* ws = (char*)d_ws;
  size_t off = 0;
  auto alloc = [&](size_t bytes) -> char* {
    off = (off + 255) & ~(size_t)255;
    char* p = ws + off;
    off += bytes;
    return p;
  };
  float*    Hs     = (float*)alloc((size_t)NPLANES * 9 * sizeof(float));
  float*    misc   = (float*)alloc(12 * sizeof(float));
  _Float16* wt0p   = (_Float16*)alloc((size_t)HIDDEN * ENC_K * sizeof(_Float16));
  _Float16* wthp   = (_Float16*)alloc((size_t)4 * HIDDEN * HIDDEN * sizeof(_Float16));
  _Float16* wtop   = (_Float16*)alloc((size_t)OUT_N * HIDDEN * sizeof(_Float16));
  _Float16* bwt0p  = (_Float16*)alloc((size_t)BH * VENC_K * sizeof(_Float16));
  _Float16* bwt1p  = (_Float16*)alloc((size_t)BH * BH * sizeof(_Float16));
  _Float16* bwtop  = (_Float16*)alloc((size_t)16 * BH * sizeof(_Float16));
  float*    biasp  = (float*)alloc((size_t)(5 * HIDDEN + OUT_N) * sizeof(float));
  float*    bbiasp = (float*)alloc((size_t)144 * sizeof(float));
  _Float16* encp   = (_Float16*)alloc((size_t)NPTS * ENC_K * sizeof(_Float16));
  _Float16* vencp  = (_Float16*)alloc((size_t)NPTS * VENC_K * sizeof(_Float16));
  float*    rgbp   = (float*)alloc((size_t)NPTS * 3 * sizeof(float));
  float*    outp   = (float*)alloc((size_t)NPTS * OUT_N * sizeof(float));
  float*    basisp = (float*)alloc((size_t)NPTS * 8 * sizeof(float));

  const int WTOT = HIDDEN * ENC_K + 4 * HIDDEN * HIDDEN + OUT_N * HIDDEN +
                   BH * VENC_K + BH * BH + 16 * BH + (5 * HIDDEN + OUT_N) + 144;
  prep_weights<<<(WTOT + 255) / 256, 256, 0, stream>>>(
      w0, b0, w1, b1, w2, b2, w3, b3, w4, b4, wo, bo,
      bw0, bb0, bw1, bb1, bwo, bbo,
      wt0p, wthp, wtop, bwt0p, bwt1p, bwtop, biasp, bbiasp);
  prep_planes<<<1, 256, 0, stream>>>(
      ref_rT, ref_t, ref_k, feat_r, feat_t, feat_center, feat_k, planes, Hs, misc);
  prep_points<<<NPTS / 256, 256, 0, stream>>>(
      Hs, misc, planes, selection, mpi_c, encp, vencp, rgbp);
  mlp_main<<<NPTS / MTILE, 256, 0, stream>>>(encp, wt0p, wthp, wtop, biasp, outp);
  mlp_basis<<<NPTS / 16, 128, 0, stream>>>(vencp, bwt0p, bwt1p, bwtop, bbiasp, basisp);
  composite<<<RAYS / 256, 256, 0, stream>>>(outp, rgbp, basisp, (float*)d_out);
}
